// ModulatedConv2d_48180943126573
// MI455X (gfx1250) — matrix operand order repaired and checked by reading, not compiled, not run on hardware
//
#include <hip/hip_runtime.h>

// ---------------- problem constants ----------------
#define NB   16      // batch
#define IC   256     // input channels
#define OC   256     // output channels
#define HH   64
#define WW   64
#define SD   512     // style dim
#define ICB  32      // input channels per LDS slab (== WMMA K)
#define EPSV 1e-8f

typedef __attribute__((ext_vector_type(16))) __bf16 v16bf;
typedef __attribute__((ext_vector_type(8)))  __bf16 v8bf;
typedef __attribute__((ext_vector_type(8)))  float  v8f;

__device__ __forceinline__ unsigned short f2bf(float f) {
    unsigned u = __builtin_bit_cast(unsigned, f);
    u += 0x7FFFu + ((u >> 16) & 1u);          // round-to-nearest-even
    return (unsigned short)(u >> 16);
}

// ---------------- s[b,i] = style[b,:] . mod_w[i,:] + mod_b[i] ----------------
__global__ void k_style(const float* __restrict__ style, const float* __restrict__ mw,
                        const float* __restrict__ mb, float* __restrict__ s) {
    int idx = blockIdx.x * blockDim.x + threadIdx.x;   // 16*256
    int b = idx >> 8, i = idx & 255;
    const float* st = style + b * SD;
    const float* w  = mw + i * SD;
    float acc = mb[i];
    for (int k = 0; k < SD; k += 4)
        acc += st[k]*w[k] + st[k+1]*w[k+1] + st[k+2]*w[k+2] + st[k+3]*w[k+3];
    s[idx] = acc;
}

// ---------------- wsq[o,i] = sum_tap w[o,i,kh,kw]^2 ----------------
__global__ void k_wsq(const float* __restrict__ w, float* __restrict__ wsq) {
    int idx = blockIdx.x * blockDim.x + threadIdx.x;   // 256*256
    const float* p = w + (size_t)idx * 9;
    float a = 0.f;
#pragma unroll
    for (int t = 0; t < 9; ++t) a += p[t] * p[t];
    wsq[idx] = a;
}

// ---------------- d[b,o] = rsqrt(sum_i wsq[o,i]*s[b,i]^2 + eps) ----------------
__global__ void k_demod(const float* __restrict__ wsq, const float* __restrict__ s,
                        float* __restrict__ d) {
    int idx = blockIdx.x * blockDim.x + threadIdx.x;   // 16*256
    int b = idx >> 8, o = idx & 255;
    const float* wo = wsq + o * IC;
    const float* sb = s + b * IC;
    float acc = 0.f;
    for (int i = 0; i < IC; ++i) { float sv = sb[i]; acc += wo[i] * sv * sv; }
    d[idx] = rsqrtf(acc + EPSV);
}

// ---------------- pack weights [o][ic][kh][kw] f32 -> [tap][o][ic] bf16 ----------------
__global__ void k_pack(const float* __restrict__ w, unsigned short* __restrict__ wp) {
    int idx = blockIdx.x * blockDim.x + threadIdx.x;   // 9*256*256
    int ic = idx & 255;
    int o  = (idx >> 8) & 255;
    int t  = idx >> 16;                                 // 0..8
    float v = w[(((size_t)o * IC + ic) * 9) + t];
    wp[((size_t)t * OC + o) * IC + ic] = f2bf(v);
}

// ---------------- main implicit-GEMM conv via v_wmma_f32_16x16x32_bf16 ----------------
// grid = (OC/64, H, B); block = 256 threads (8 wave32).
// Block tile: 64 oc x 64 w-positions at fixed (b,h). Wave tile: 16 oc x 32 w.
#define XS_ELEMS (3 * 66 * ICB)   // 3 input rows x 66 cols (halo) x 32 ic = 6336 bf16

__global__ __launch_bounds__(256) void k_conv(
    const float* __restrict__ x, const float* __restrict__ s,
    const float* __restrict__ dmd, const unsigned short* __restrict__ wpu,
    float* __restrict__ out) {

    __shared__ __align__(32) unsigned short xs[XS_ELEMS];   // [row][col][ic], 12.4 KB
    const __bf16* wp = reinterpret_cast<const __bf16*>(wpu);

    const int tid   = threadIdx.x;
    const int lane  = tid & 31;
    const int wv    = tid >> 5;
    const int msub  = wv & 3;         // which 16-row oc sub-tile
    const int npair = wv >> 2;        // which 32-col n half
    const int b   = blockIdx.z;
    const int h   = blockIdx.y;
    const int oc0 = blockIdx.x * 64;

    const int m    = lane & 15;       // A-row / B-col / D-col index for this lane
    const int hgrp = lane >> 4;       // lane half (K grouping per ISA layout)
    const int nb0  = npair * 32;      // first 16-col sub-tile base
    const int nb1  = nb0 + 16;        // second

    v8f c0 = {};
    v8f c1 = {};
    const float* sb = s + b * IC;

    for (int ic0 = 0; ic0 < IC; ic0 += ICB) {
        __syncthreads();
        // stage modulated input slab to LDS as bf16: xs[r][col][ic], halo + zero pad
        for (int e = tid; e < XS_ELEMS; e += 256) {
            int ic = e & (ICB - 1);
            int c  = (e >> 5) % 66;
            int r  = e / (66 * ICB);
            int gh = h + r - 1, gw = c - 1;
            float v = 0.f;
            if ((unsigned)gh < HH && (unsigned)gw < WW)
                v = x[((size_t)(b * IC + ic0 + ic) * HH + gh) * WW + gw] * sb[ic0 + ic];
            xs[e] = f2bf(v);
        }
        __syncthreads();

#pragma unroll
        for (int t = 0; t < 9; ++t) {
            const int kh = t / 3, kw = t - kh * 3;

            // A fragment: 16x32 bf16 weight tile, rows = oc, K = ic.
            // A layout (matching B): lane half 0 holds K 0-15 (v0-7);
            // half 1 holds K 16-31. Packed weights give contiguous K.
            size_t wbase = ((size_t)(t * OC + oc0 + msub * 16 + m)) * IC + ic0;
            const v8bf alo = *reinterpret_cast<const v8bf*>(wp + wbase + (hgrp ? 16 : 0));
            const v8bf ahi = *reinterpret_cast<const v8bf*>(wp + wbase + (hgrp ? 24 : 8));
            v16bf a;
#pragma unroll
            for (int i = 0; i < 8; ++i) { a[i] = alo[i]; a[8 + i] = ahi[i]; }

            // B fragments: 32x16, column = lane%16, K = ic contiguous in LDS ->
            // one 32-byte LDS load per fragment (lanes 0-15: K0-15, 16-31: K16-31).
            const int col0 = nb0 + m + kw;   // 0..65 (halo handles kw-1 shift)
            const int col1 = nb1 + m + kw;
            const v16bf bf0 = *reinterpret_cast<const v16bf*>(&xs[(kh * 66 + col0) * ICB + hgrp * 16]);
            const v16bf bf1 = *reinterpret_cast<const v16bf*>(&xs[(kh * 66 + col1) * ICB + hgrp * 16]);

            c0 = __builtin_amdgcn_wmma_f32_16x16x32_bf16(false, a, false, bf0, (short)0, c0, false, false);
            c1 = __builtin_amdgcn_wmma_f32_16x16x32_bf16(false, a, false, bf1, (short)0, c1, false, false);
        }
    }

    // D layout: vgpr v -> oc row v + (lane<16 ? 0 : 8); col = lane%16. Demodulate at store.
#pragma unroll
    for (int v = 0; v < 8; ++v) {
        int oc = oc0 + msub * 16 + hgrp * 8 + v;
        float ds = dmd[b * OC + oc];
        size_t rowbase = ((size_t)(b * OC + oc) * HH + h) * WW;
        out[rowbase + nb0 + m] = c0[v] * ds;
        out[rowbase + nb1 + m] = c1[v] * ds;
    }
}

// ---------------- launcher ----------------
extern "C" void kernel_launch(void* const* d_in, const int* in_sizes, int n_in,
                              void* d_out, int out_size, void* d_ws, size_t ws_size,
                              hipStream_t stream) {
    const float* x      = (const float*)d_in[0];
    const float* style  = (const float*)d_in[1];
    const float* weight = (const float*)d_in[2];
    const float* mod_w  = (const float*)d_in[3];
    const float* mod_b  = (const float*)d_in[4];
    float* out = (float*)d_out;

    // workspace layout (all offsets 256B-aligned): ~1.45 MB total
    char* ws = (char*)d_ws;
    float*          s_buf = (float*)(ws);                         // 16 KB
    float*          d_buf = (float*)(ws + (16 << 10));            // 16 KB
    float*          wsq   = (float*)(ws + (32 << 10));            // 256 KB
    unsigned short* wp    = (unsigned short*)(ws + (32 << 10) + (256 << 10)); // 1.18 MB

    k_style<<<(NB * IC) / 256, 256, 0, stream>>>(style, mod_w, mod_b, s_buf);
    k_wsq  <<<(OC * IC) / 256, 256, 0, stream>>>(weight, wsq);
    k_demod<<<(NB * OC) / 256, 256, 0, stream>>>(wsq, s_buf, d_buf);
    k_pack <<<(9 * OC * IC) / 256, 256, 0, stream>>>(weight, wp);

    dim3 grid(OC / 64, HH, NB);
    k_conv<<<grid, 256, 0, stream>>>(x, s_buf, d_buf, wp, out);
}
